// HardwareVMMoperation_43508018708705
// MI455X (gfx1250) — matrix operand order repaired and checked by reading, not compiled, not run on hardware
//
#include <hip/hip_runtime.h>

typedef _Float16 v16h __attribute__((ext_vector_type(16)));
typedef _Float16 v8h  __attribute__((ext_vector_type(8)));
typedef _Float16 h4   __attribute__((ext_vector_type(4)));
typedef float    v8f  __attribute__((ext_vector_type(8)));

#define TILE_M 128
#define TILE_N 128
#define MAXC   2.56f
#define QSCALE 255.0f

// ---------------- f16 fast path: KSTEP=64, pitch 72 halfwords ----------------
#define KS2    64
#define PITCH2 72                       // 64 data + 8 pad (144B rows -> conflict-free b128)
#define BUFH2  (TILE_M * PITCH2)        // f16 elements per buffer per matrix

__device__ __forceinline__ void async_ld_b128(void* lds_ptr, const void* gptr) {
    // generic pointers to __shared__ keep the LDS byte offset in bits [31:0]
    unsigned loff = (unsigned)(size_t)lds_ptr;
    asm volatile("global_load_async_to_lds_b128 %0, %1, off"
                 :: "v"(loff), "v"(gptr) : "memory");
}
__device__ __forceinline__ void wait_async0() {
    asm volatile("s_wait_asynccnt 0x0" ::: "memory");
}

__global__ __launch_bounds__(256)
void hard_fc_wmma_f16_async(const _Float16* __restrict__ X,   // [B,K] f16
                            const _Float16* __restrict__ W,   // [N,K] f16
                            const float* __restrict__ bias,
                            float* __restrict__ out,
                            int K, int N, int nsteps)
{
    __shared__ _Float16 ldsA[2][BUFH2];
    __shared__ _Float16 ldsB[2][BUFH2];

    const int tid  = threadIdx.x;
    const int lane = tid & 31;
    const int wv   = tid >> 5;
    const int wm   = wv >> 1;          // 0..3 (M)
    const int wn   = wv & 1;           // 0..1 (N)
    const int hs   = lane >> 4;
    const int lr   = lane & 15;

    const int blockM = blockIdx.y * TILE_M;
    const int blockN = blockIdx.x * TILE_N;

    // staging: tile = 128 rows x 64 f16 = 16KB = 1024 x 16B chunks; 4 per thread per matrix
    const _Float16* gA[4];
    const _Float16* gB[4];
    _Float16* lA0[4];
    _Float16* lB0[4];
#pragma unroll
    for (int i = 0; i < 4; ++i) {
        int q   = tid + i * 256;       // 0..1023
        int row = q >> 3;              // 0..127
        int ch  = q & 7;               // 16B chunk within 64-f16 row
        gA[i]  = X + (size_t)(blockM + row) * K + ch * 8;
        gB[i]  = W + (size_t)(blockN + row) * K + ch * 8;
        lA0[i] = &ldsA[0][row * PITCH2 + ch * 8];
        lB0[i] = &ldsB[0][row * PITCH2 + ch * 8];
    }

    const v8f vzero = {};
    v8f c[8];       // per-128-chunk partial accumulators
    v8f acc[8];     // persistent quantized accumulators
#pragma unroll
    for (int t = 0; t < 8; ++t) { c[t] = vzero; acc[t] = vzero; }

    // prologue: async-stage step 0 into buffer 0
#pragma unroll
    for (int i = 0; i < 4; ++i) {
        async_ld_b128(lA0[i], gA[i]);
        async_ld_b128(lB0[i], gB[i]);
    }
    wait_async0();
    __syncthreads();

    for (int s = 0; s < nsteps; ++s) {
        const int cur = s & 1;

        // async-stage next K-step into the other buffer (DMA, no VALU/regs)
        if (s + 1 < nsteps) {
            const size_t go = (size_t)(s + 1) * KS2;
            const size_t lo = (size_t)(cur ^ 1) * BUFH2;
#pragma unroll
            for (int i = 0; i < 4; ++i) {
                async_ld_b128(lA0[i] + lo, gA[i] + go);
                async_ld_b128(lB0[i] + lo, gB[i] + go);
            }
        }

        // 2 x (K=32) WMMA slices out of the 64-deep buffer
#pragma unroll
        for (int kk = 0; kk < 2; ++kk) {
            v16h af[2];
#pragma unroll
            for (int mt = 0; mt < 2; ++mt) {
                const int idx = (wm * 32 + mt * 16 + lr) * PITCH2 + kk * 32 + hs * 8;
                v8h lo = *(const v8h*)&ldsA[cur][idx];
                v8h hi = *(const v8h*)&ldsA[cur][idx + 16];
                af[mt] = __builtin_shufflevector(lo, hi, 0,1,2,3,4,5,6,7,8,9,10,11,12,13,14,15);
            }
            v16h bf[4];
#pragma unroll
            for (int nt = 0; nt < 4; ++nt) {
                const int idx = (wn * 64 + nt * 16 + lr) * PITCH2 + kk * 32 + hs * 8;
                v8h lo = *(const v8h*)&ldsB[cur][idx];
                v8h hi = *(const v8h*)&ldsB[cur][idx + 16];
                bf[nt] = __builtin_shufflevector(lo, hi, 0,1,2,3,4,5,6,7,8,9,10,11,12,13,14,15);
            }
#pragma unroll
            for (int mt = 0; mt < 2; ++mt)
#pragma unroll
                for (int nt = 0; nt < 4; ++nt) {
                    const int t = mt * 4 + nt;
                    c[t] = __builtin_amdgcn_wmma_f32_16x16x32_f16(
                        false, af[mt], false, bf[nt], (short)0, c[t], false, false);
                }
        }

        // end of 128-deep chunk (2 steps): ADC quantization
        if (s & 1) {
#pragma unroll
            for (int t = 0; t < 8; ++t) {
#pragma unroll
                for (int e = 0; e < 8; ++e) {
                    float y = c[t][e];
                    y = fminf(fmaxf(y, -MAXC), MAXC);
                    y = __builtin_rintf(y * (QSCALE / MAXC));
                    acc[t][e] += y * (MAXC / QSCALE);
                }
                c[t] = vzero;
            }
        }

        wait_async0();
        __syncthreads();
    }

    // epilogue: bias + store (C layout: M = e + 8*hs, N = lr)
#pragma unroll
    for (int mt = 0; mt < 2; ++mt) {
#pragma unroll
        for (int nt = 0; nt < 4; ++nt) {
            const int   t  = mt * 4 + nt;
            const int   n  = blockN + wn * 64 + nt * 16 + lr;
            const float bv = bias[n];
            const int   m0 = blockM + wm * 32 + mt * 16 + hs * 8;
#pragma unroll
            for (int e = 0; e < 8; ++e)
                out[(size_t)(m0 + e) * N + n] = acc[t][e] + bv;
        }
    }
}

// ---------------- f32 -> f16 pre-conversion (memory-bound) ----------------
__global__ __launch_bounds__(256)
void cvt_f32_f16_kernel(const float4* __restrict__ src, h4* __restrict__ dst, long long n4)
{
    long long i      = (long long)blockIdx.x * blockDim.x + threadIdx.x;
    long long stride = (long long)gridDim.x * blockDim.x;
    for (; i < n4; i += stride) {
        float4 v = src[i];
        h4 h = { (_Float16)v.x, (_Float16)v.y, (_Float16)v.z, (_Float16)v.w };
        dst[i] = h;
    }
}

// ---------------- fallback: fused f32 staging path (round-1 kernel) ----------------
#define KSTEP  32
#define PITCH  40

__global__ __launch_bounds__(256)
void hard_fc_wmma_kernel(const float* __restrict__ X,
                         const float* __restrict__ W,
                         const float* __restrict__ bias,
                         float* __restrict__ out,
                         int K, int N, int nsteps)
{
    __shared__ _Float16 ldsA[2][TILE_M * PITCH];
    __shared__ _Float16 ldsB[2][TILE_N * PITCH];

    const int tid  = threadIdx.x;
    const int lane = tid & 31;
    const int wv   = tid >> 5;
    const int wm   = wv >> 1;
    const int wn   = wv & 1;
    const int hs   = lane >> 4;
    const int lr   = lane & 15;

    const int blockM = blockIdx.y * TILE_M;
    const int blockN = blockIdx.x * TILE_N;

    const float4* pA4[4];
    const float4* pB4[4];
    int stOff[4];
#pragma unroll
    for (int i = 0; i < 4; ++i) {
        int q   = tid + i * 256;
        int row = q >> 3;
        int cg  = q & 7;
        pA4[i]   = (const float4*)(X + (size_t)(blockM + row) * K + cg * 4);
        pB4[i]   = (const float4*)(W + (size_t)(blockN + row) * K + cg * 4);
        stOff[i] = row * PITCH + cg * 4;
    }

    int offA[2], offB[4];
#pragma unroll
    for (int mt = 0; mt < 2; ++mt)
        offA[mt] = (wm * 32 + mt * 16 + lr) * PITCH + hs * 8;
#pragma unroll
    for (int nt = 0; nt < 4; ++nt)
        offB[nt] = (wn * 64 + nt * 16 + lr) * PITCH + hs * 8;

    const v8f vzero = {};
    v8f c[8];
    v8f acc[8];
#pragma unroll
    for (int t = 0; t < 8; ++t) { c[t] = vzero; acc[t] = vzero; }

#pragma unroll
    for (int i = 0; i < 4; ++i) {
        float4 a = pA4[i][0];
        float4 b = pB4[i][0];
        h4 ha = { (_Float16)a.x, (_Float16)a.y, (_Float16)a.z, (_Float16)a.w };
        h4 hb = { (_Float16)b.x, (_Float16)b.y, (_Float16)b.z, (_Float16)b.w };
        *(h4*)&ldsA[0][stOff[i]] = ha;
        *(h4*)&ldsB[0][stOff[i]] = hb;
    }
    __syncthreads();

    for (int s = 0; s < nsteps; ++s) {
        const int  cur     = s & 1;
        const bool hasNext = (s + 1) < nsteps;

        float4 ra[4], rb[4];
        if (hasNext) {
#pragma unroll
            for (int i = 0; i < 4; ++i) {
                ra[i] = pA4[i][(size_t)(s + 1) * 8];
                rb[i] = pB4[i][(size_t)(s + 1) * 8];
            }
        }

        v16h af[2];
#pragma unroll
        for (int mt = 0; mt < 2; ++mt) {
            v8h lo = *(const v8h*)&ldsA[cur][offA[mt]];
            v8h hi = *(const v8h*)&ldsA[cur][offA[mt] + 16];
            af[mt] = __builtin_shufflevector(lo, hi, 0,1,2,3,4,5,6,7,8,9,10,11,12,13,14,15);
        }
        v16h bf[4];
#pragma unroll
        for (int nt = 0; nt < 4; ++nt) {
            v8h lo = *(const v8h*)&ldsB[cur][offB[nt]];
            v8h hi = *(const v8h*)&ldsB[cur][offB[nt] + 16];
            bf[nt] = __builtin_shufflevector(lo, hi, 0,1,2,3,4,5,6,7,8,9,10,11,12,13,14,15);
        }

#pragma unroll
        for (int mt = 0; mt < 2; ++mt)
#pragma unroll
            for (int nt = 0; nt < 4; ++nt) {
                const int t = mt * 4 + nt;
                c[t] = __builtin_amdgcn_wmma_f32_16x16x32_f16(
                    false, af[mt], false, bf[nt], (short)0, c[t], false, false);
            }

        if (hasNext) {
            const int nxt = cur ^ 1;
#pragma unroll
            for (int i = 0; i < 4; ++i) {
                h4 ha = { (_Float16)ra[i].x, (_Float16)ra[i].y, (_Float16)ra[i].z, (_Float16)ra[i].w };
                h4 hb = { (_Float16)rb[i].x, (_Float16)rb[i].y, (_Float16)rb[i].z, (_Float16)rb[i].w };
                *(h4*)&ldsA[nxt][stOff[i]] = ha;
                *(h4*)&ldsB[nxt][stOff[i]] = hb;
            }
        }

        if ((s & 3) == 3) {
#pragma unroll
            for (int t = 0; t < 8; ++t) {
#pragma unroll
                for (int e = 0; e < 8; ++e) {
                    float y = c[t][e];
                    y = fminf(fmaxf(y, -MAXC), MAXC);
                    y = __builtin_rintf(y * (QSCALE / MAXC));
                    acc[t][e] += y * (MAXC / QSCALE);
                }
                c[t] = vzero;
            }
        }
        __syncthreads();
    }

#pragma unroll
    for (int mt = 0; mt < 2; ++mt) {
#pragma unroll
        for (int nt = 0; nt < 4; ++nt) {
            const int   t  = mt * 4 + nt;
            const int   n  = blockN + wn * 64 + nt * 16 + lr;
            const float bv = bias[n];
            const int   m0 = blockM + wm * 32 + mt * 16 + hs * 8;
#pragma unroll
            for (int e = 0; e < 8; ++e)
                out[(size_t)(m0 + e) * N + n] = acc[t][e] + bv;
        }
    }
}

extern "C" void kernel_launch(void* const* d_in, const int* in_sizes, int n_in,
                              void* d_out, int out_size, void* d_ws, size_t ws_size,
                              hipStream_t stream) {
    const float* X    = (const float*)d_in[0];   // [B, K] f32
    const float* Wt   = (const float*)d_in[1];   // [N, K] f32
    const float* bias = (const float*)d_in[2];   // [N] f32
    float*       out  = (float*)d_out;           // [B, N] f32

    const int N  = in_sizes[2];
    const int K  = in_sizes[1] / N;
    const int Bm = in_sizes[0] / K;

    dim3 grid(N / TILE_N, Bm / TILE_M);
    dim3 block(256);

    const size_t needX = (size_t)Bm * K * sizeof(_Float16);
    const size_t needW = (size_t)N  * K * sizeof(_Float16);

    if (ws_size >= needX + needW && (K % 128) == 0) {
        // fast path: pre-convert to f16, then async-staged WMMA GEMM
        _Float16* X16 = (_Float16*)d_ws;
        _Float16* W16 = (_Float16*)((char*)d_ws + needX);
        cvt_f32_f16_kernel<<<4096, 256, 0, stream>>>(
            (const float4*)X, (h4*)X16, (long long)Bm * K / 4);
        cvt_f32_f16_kernel<<<2048, 256, 0, stream>>>(
            (const float4*)Wt, (h4*)W16, (long long)N * K / 4);
        hard_fc_wmma_f16_async<<<grid, block, 0, stream>>>(
            X16, W16, bias, out, K, N, K / KS2);
    } else {
        // fallback: fused f32->f16 staging
        hard_fc_wmma_kernel<<<grid, block, 0, stream>>>(
            X, Wt, bias, out, K, N, K / KSTEP);
    }
}
